// CausalSelfAttention_3667902071444
// MI455X (gfx1250) — matrix operand order repaired and checked by reading, not compiled, not run on hardware
//
#include <hip/hip_runtime.h>

#define BATCH 2
#define SEQ   2048
#define EMB   1024
#define NH    16
#define HD    64

#define USE_ASYNC_LDS 1

// padded LDS row strides (in bf16 elements) to spread DS banks
#define BROW 40   // 32-elem rows + 16B pad  -> 80B stride
#define KROW 72   // 64-elem rows + 16B pad  -> 144B stride

typedef __bf16 bf16_t;
typedef __attribute__((ext_vector_type(16))) __bf16 v16bf;
typedef __attribute__((ext_vector_type(8)))  float  v8f;

union Frag { v16bf v; uint4 q[2]; };

__device__ __forceinline__ v8f wmma_bf16(v16bf a, v16bf b, v8f c) {
  return __builtin_amdgcn_wmma_f32_16x16x32_bf16(false, a, false, b, (short)0, c, false, false);
}

// Scheduling fence: keep the batched DS loads above the WMMA chain so the
// waitcnt pass emits staggered s_wait_dscnt instead of 0x0 per WMMA.
__device__ __forceinline__ void sched_fence() {
#if defined(__has_builtin) && __has_builtin(__builtin_amdgcn_sched_barrier)
  __builtin_amdgcn_sched_barrier(0);
#endif
}

// A-matrix 16x32 bf16 fragment (ISA 7.12.2): lane holds row M=lane%16,
// K bytes [16*hi, +16) and [32+16*hi, +16) within a 64-byte K-step.
__device__ __forceinline__ void load_a(Frag& f, const bf16_t* row, int kbyte, int hi) {
  const char* p = (const char*)row + kbyte;
  f.q[0] = *(const uint4*)(p + hi * 16);
  f.q[1] = *(const uint4*)(p + 32 + hi * 16);
}

// B-matrix 32x16 bf16 fragment: lane holds column N=lane%16 (row of B^T),
// K bytes [16*hi, +16) and [32+16*hi, +16) within a 64-byte K-step (same as A).
__device__ __forceinline__ void load_b(Frag& f, const bf16_t* rowT, int kbyte, int hi) {
  const char* p = (const char*)rowT + kbyte;
  f.q[0] = *(const uint4*)(p + hi * 16);
  f.q[1] = *(const uint4*)(p + 32 + hi * 16);
}

__device__ __forceinline__ bf16_t f2bf(float x) {
  union { float f; unsigned u; } c; c.f = x;
  unsigned r = c.u + 0x7FFFu + ((c.u >> 16) & 1u);
  union { unsigned short s; bf16_t b; } o; o.s = (unsigned short)(r >> 16);
  return o.b;
}

// 16-byte global -> LDS copy via CDNA5 ASYNCcnt DMA
// (GLOBAL_LOAD_ASYNC_TO_LDS_B128). The clang builtin takes
// (v4i AS(1)* src, v4i AS(3)* dst, int offset, int cpol); with it the waitcnt
// pass models ASYNCcnt precisely. LDS dst = low 32 bits of the generic shared
// pointer (ISA 10.2 aperture layout).
#if USE_ASYNC_LDS && defined(__has_builtin)
#if __has_builtin(__builtin_amdgcn_global_load_async_to_lds_b128)
#define HAVE_ASYNC_BUILTIN 1
#endif
#endif

typedef int v4i_vs __attribute__((vector_size(16)));
typedef __attribute__((address_space(1))) v4i_vs* as1_v4i_ptr;
typedef __attribute__((address_space(3))) v4i_vs* as3_v4i_ptr;

__device__ __forceinline__ void cp16(void* l, const void* g) {
#if USE_ASYNC_LDS
#if defined(HAVE_ASYNC_BUILTIN)
  __builtin_amdgcn_global_load_async_to_lds_b128(
      (as1_v4i_ptr)(unsigned long long)g,
      (as3_v4i_ptr)(unsigned)(unsigned long long)l,
      0, 0);
#else
  unsigned lo = (unsigned)(unsigned long long)l;
  asm volatile("global_load_async_to_lds_b128 %0, %1, off"
               :: "v"(lo), "v"(g) : "memory");
#endif
#else
  *(uint4*)l = *(const uint4*)g;
#endif
}

__device__ __forceinline__ void wait_async() {
#if USE_ASYNC_LDS
#if defined(__has_builtin) && __has_builtin(__builtin_amdgcn_s_wait_asynccnt)
  __builtin_amdgcn_s_wait_asynccnt(0);
#else
  asm volatile("s_wait_asynccnt 0x0" ::: "memory");
#endif
#endif
}

// ---------------- conversion / transpose kernels ----------------

__global__ __launch_bounds__(256) void cvt_f32_bf16(const float* __restrict__ in,
                                                    bf16_t* __restrict__ out, int n) {
  int i = blockIdx.x * 256 + threadIdx.x;
  if (i < n) out[i] = f2bf(in[i]);
}

// in: [R][C] f32  ->  out: [C][R] bf16
__global__ __launch_bounds__(256) void transpose_bf16(const float* __restrict__ in,
                                                      bf16_t* __restrict__ out,
                                                      int R, int C) {
  int i = blockIdx.x * 256 + threadIdx.x;
  if (i < R * C) {
    int orow = i / R;
    int ocol = i - orow * R;
    out[(size_t)orow * R + ocol] = f2bf(in[(size_t)ocol * C + orow]);
  }
}

// ---------------- QKV GEMM: [B*T,E] x [E,3E] -> Q,K (B*H,T,D), V^T (B*H,D,T) ----------------
// Block: 8 waves, tile 128(M) x 128(N), K-step 32. B-tile double-buffered in
// LDS via async DMA; A fragment double-buffered in registers.

__global__ __launch_bounds__(256) void qkv_gemm(const bf16_t* __restrict__ X,
                                                const bf16_t* __restrict__ WT,   // [3E][E]
                                                bf16_t* __restrict__ Qo,
                                                bf16_t* __restrict__ Ko,
                                                bf16_t* __restrict__ VTo) {
  __shared__ alignas(16) bf16_t bsm[2][128 * BROW];   // [n][k] padded rows
  const int lane = threadIdx.x & 31;
  const int wave = threadIdx.x >> 5;
  const int hi = lane >> 4;
  const int lr = lane & 15;
  const int m0 = blockIdx.x * 128 + wave * 16;
  const int n0 = blockIdx.y * 128;

  const int srow = threadIdx.x >> 1;    // 0..127 staged B row
  const int shalf = threadIdx.x & 1;    // 32B half of the 64B row payload
  const bf16_t* bsrc = WT + (size_t)(n0 + srow) * EMB + shalf * 16;

  v8f acc[8] = {};
  const bf16_t* arow = X + (size_t)(m0 + lr) * EMB;

  {
    void* l = &bsm[0][srow * BROW + shalf * 16];
    cp16(l, bsrc);
    cp16((char*)l + 16, (const char*)bsrc + 16);
  }
  Frag a_cur, a_nxt;
  load_a(a_cur, arow, 0, hi);
  a_nxt = a_cur;

  int p = 0;
  for (int k0 = 0; k0 < EMB; k0 += 32) {
    wait_async();
    __syncthreads();
    if (k0 + 32 < EMB) {
      void* l = &bsm[p ^ 1][srow * BROW + shalf * 16];
      const bf16_t* g = bsrc + (k0 + 32);
      cp16(l, g);
      cp16((char*)l + 16, (const char*)g + 16);
      load_a(a_nxt, arow, 2 * (k0 + 32), hi);
    }
    // batch all B-fragment LDS loads, fence, then the WMMA chain
    Frag b[8];
#pragma unroll
    for (int j = 0; j < 8; ++j) load_b(b[j], &bsm[p][(j * 16 + lr) * BROW], 0, hi);
    sched_fence();
#pragma unroll
    for (int j = 0; j < 8; ++j) acc[j] = wmma_bf16(a_cur.v, b[j].v, acc[j]);
    a_cur = a_nxt;
    p ^= 1;
  }

  const float qscale = 0.125f;  // 1/sqrt(HD)
#pragma unroll
  for (int j = 0; j < 8; ++j) {
    const int n = n0 + j * 16 + lr;
#pragma unroll
    for (int r = 0; r < 8; ++r) {
      const int m = m0 + r + 8 * hi;
      const int bb = m >> 11;
      const int t = m & (SEQ - 1);
      const float val = acc[j][r];
      if (n < EMB) {
        const int h = n >> 6, d = n & 63;
        Qo[(((size_t)(bb * NH + h)) * SEQ + t) * HD + d] = f2bf(val * qscale);
      } else if (n < 2 * EMB) {
        const int nn = n - EMB;
        const int h = nn >> 6, d = nn & 63;
        Ko[(((size_t)(bb * NH + h)) * SEQ + t) * HD + d] = f2bf(val);
      } else {
        const int nn = n - 2 * EMB;
        const int h = nn >> 6, d = nn & 63;
        VTo[(((size_t)(bb * NH + h)) * HD + d) * SEQ + t] = f2bf(val);
      }
    }
  }
}

// ---------------- flash attention: one block = (b,h, 64 q-rows), 4 waves ----------------

__global__ __launch_bounds__(128) void flash_attn(const bf16_t* __restrict__ Q,
                                                  const bf16_t* __restrict__ K,
                                                  const bf16_t* __restrict__ VT,
                                                  bf16_t* __restrict__ O) {
  __shared__ alignas(16) bf16_t kbuf[2][64 * KROW];   // [t][d], padded rows
  __shared__ alignas(16) bf16_t vbuf[2][64 * KROW];   // [d][t], padded rows
  __shared__ alignas(16) bf16_t pbuf[4 * 16 * KROW];  // per-wave P staging

  const int lane = threadIdx.x & 31;
  const int wave = threadIdx.x >> 5;
  const int hi = lane >> 4;
  const int lr = lane & 15;
  const int qt = blockIdx.x & 31;
  const int bh = blockIdx.x >> 5;
  const size_t base = (size_t)bh * SEQ * HD;          // same for Q,K and V^T
  const int q0 = qt * 64 + wave * 16;

  Frag qf[2];
  {
    const bf16_t* qrow = Q + base + (size_t)(q0 + lr) * HD;
    load_a(qf[0], qrow, 0, hi);
    load_a(qf[1], qrow, 64, hi);
  }

  float rm[8], rl[8];
  v8f o[4] = {};
#pragma unroll
  for (int r = 0; r < 8; ++r) { rm[r] = -1e30f; rl[r] = 0.0f; }

  const int srow = threadIdx.x >> 1;   // 64 rows, 64B (half-row) per thread per tile
  const int shalf = threadIdx.x & 1;

  auto stage_bytes = [&](int kt, int pp) {
    const int kbase = kt * 64;
    const char* ks = (const char*)(K + base + (size_t)(kbase + srow) * HD + shalf * 32);
    const char* vs = (const char*)(VT + base + (size_t)srow * SEQ + kbase + shalf * 32);
    char* kl = (char*)&kbuf[pp][srow * KROW + shalf * 32];
    char* vl = (char*)&vbuf[pp][srow * KROW + shalf * 32];
#pragma unroll
    for (int c = 0; c < 4; ++c) {
      cp16(kl + c * 16, ks + c * 16);
      cp16(vl + c * 16, vs + c * 16);
    }
  };

  stage_bytes(0, 0);
  int p = 0;
  for (int kt = 0; kt <= qt; ++kt) {
    wait_async();
    __syncthreads();
    if (kt + 1 <= qt) stage_bytes(kt + 1, p ^ 1);

    // S = Q * K^T : per-wave 16x64, Kdim = HD (2 steps of 32)
    v8f s[4] = {};
    {
      Frag b[8];
#pragma unroll
      for (int ss = 0; ss < 2; ++ss)
#pragma unroll
        for (int j = 0; j < 4; ++j)
          load_b(b[ss * 4 + j], &kbuf[p][(j * 16 + lr) * KROW], ss * 64, hi);
      sched_fence();
#pragma unroll
      for (int ss = 0; ss < 2; ++ss)
#pragma unroll
        for (int j = 0; j < 4; ++j)
          s[j] = wmma_bf16(qf[ss].v, b[ss * 4 + j].v, s[j]);
    }

    if (kt == qt) {  // causal mask on diagonal tile
#pragma unroll
      for (int j = 0; j < 4; ++j) {
        const int krel = j * 16 + lr;
#pragma unroll
        for (int r = 0; r < 8; ++r) {
          const int qrel = wave * 16 + r + 8 * hi;
          s[j][r] = (krel > qrel) ? -1e30f : s[j][r];
        }
      }
    }

    // online softmax (rows live across 16 lanes of a half-wave)
#pragma unroll
    for (int r = 0; r < 8; ++r) {
      float mt = fmaxf(fmaxf(s[0][r], s[1][r]), fmaxf(s[2][r], s[3][r]));
#pragma unroll
      for (int off = 8; off >= 1; off >>= 1) mt = fmaxf(mt, __shfl_xor(mt, off, 32));
      const float mnew = fmaxf(rm[r], mt);
      const float alpha = __expf(rm[r] - mnew);
      float ps = 0.0f;
#pragma unroll
      for (int j = 0; j < 4; ++j) { const float pv = __expf(s[j][r] - mnew); s[j][r] = pv; ps += pv; }
#pragma unroll
      for (int off = 8; off >= 1; off >>= 1) ps += __shfl_xor(ps, off, 32);
      rl[r] = rl[r] * alpha + ps;
      rm[r] = mnew;
#pragma unroll
      for (int j = 0; j < 4; ++j) o[j][r] *= alpha;
    }

    // P: C-layout -> per-wave LDS row-major -> reload as A fragments
    bf16_t* pw = pbuf + wave * (16 * KROW);
#pragma unroll
    for (int j = 0; j < 4; ++j) {
#pragma unroll
      for (int r = 0; r < 8; ++r) {
        pw[(r + 8 * hi) * KROW + j * 16 + lr] = f2bf(s[j][r]);
      }
    }
    Frag pf[2];
    load_a(pf[0], pw + lr * KROW, 0, hi);
    load_a(pf[1], pw + lr * KROW, 64, hi);

    // O += P * V  (Kdim = 64 keys, B^T rows = V^T rows, contiguous)
    {
      Frag vf[8];
#pragma unroll
      for (int ss = 0; ss < 2; ++ss)
#pragma unroll
        for (int j = 0; j < 4; ++j)
          load_b(vf[ss * 4 + j], &vbuf[p][(j * 16 + lr) * KROW], ss * 64, hi);
      sched_fence();
#pragma unroll
      for (int ss = 0; ss < 2; ++ss)
#pragma unroll
        for (int j = 0; j < 4; ++j)
          o[j] = wmma_bf16(pf[ss].v, vf[ss * 4 + j].v, o[j]);
    }
    p ^= 1;
  }

  const int bb = bh >> 4, h = bh & 15;
#pragma unroll
  for (int r = 0; r < 8; ++r) {
    const float inv = 1.0f / rl[r];
    const int t = q0 + r + 8 * hi;
#pragma unroll
    for (int j = 0; j < 4; ++j) {
      O[((size_t)bb * SEQ + t) * EMB + h * 64 + j * 16 + lr] = f2bf(o[j][r] * inv);
    }
  }
}

// ---------------- projection GEMM: [B*T,E] x [E,E] + bias -> f32 out ----------------

__global__ __launch_bounds__(256) void proj_gemm(const bf16_t* __restrict__ A,
                                                 const bf16_t* __restrict__ WT,  // [E][E]
                                                 const float* __restrict__ bias,
                                                 float* __restrict__ out) {
  __shared__ alignas(16) bf16_t bsm[2][128 * BROW];
  const int lane = threadIdx.x & 31;
  const int wave = threadIdx.x >> 5;
  const int hi = lane >> 4;
  const int lr = lane & 15;
  const int m0 = blockIdx.x * 128 + wave * 16;
  const int n0 = blockIdx.y * 128;

  const int srow = threadIdx.x >> 1;
  const int shalf = threadIdx.x & 1;
  const bf16_t* bsrc = WT + (size_t)(n0 + srow) * EMB + shalf * 16;

  v8f acc[8] = {};
  const bf16_t* arow = A + (size_t)(m0 + lr) * EMB;

  {
    void* l = &bsm[0][srow * BROW + shalf * 16];
    cp16(l, bsrc);
    cp16((char*)l + 16, (const char*)bsrc + 16);
  }
  Frag a_cur, a_nxt;
  load_a(a_cur, arow, 0, hi);
  a_nxt = a_cur;

  int p = 0;
  for (int k0 = 0; k0 < EMB; k0 += 32) {
    wait_async();
    __syncthreads();
    if (k0 + 32 < EMB) {
      void* l = &bsm[p ^ 1][srow * BROW + shalf * 16];
      const bf16_t* g = bsrc + (k0 + 32);
      cp16(l, g);
      cp16((char*)l + 16, (const char*)g + 16);
      load_a(a_nxt, arow, 2 * (k0 + 32), hi);
    }
    Frag b[8];
#pragma unroll
    for (int j = 0; j < 8; ++j) load_b(b[j], &bsm[p][(j * 16 + lr) * BROW], 0, hi);
    sched_fence();
#pragma unroll
    for (int j = 0; j < 8; ++j) acc[j] = wmma_bf16(a_cur.v, b[j].v, acc[j]);
    a_cur = a_nxt;
    p ^= 1;
  }

#pragma unroll
  for (int j = 0; j < 8; ++j) {
    const int n = n0 + j * 16 + lr;
    const float bv = bias[n];
#pragma unroll
    for (int r = 0; r < 8; ++r) {
      const int m = m0 + r + 8 * hi;
      out[(size_t)m * EMB + n] = acc[j][r] + bv;
    }
  }
}

// ---------------- host launcher ----------------

extern "C" void kernel_launch(void* const* d_in, const int* in_sizes, int n_in,
                              void* d_out, int out_size, void* d_ws, size_t ws_size,
                              hipStream_t stream) {
  (void)in_sizes; (void)n_in; (void)out_size; (void)ws_size;
  const float* x      = (const float*)d_in[0];
  const float* w_qkv  = (const float*)d_in[1];
  const float* w_proj = (const float*)d_in[2];
  const float* b_proj = (const float*)d_in[3];
  float* out = (float*)d_out;

  char* ws = (char*)d_ws;
  bf16_t* xb     = (bf16_t*)ws; ws += (size_t)BATCH * SEQ * EMB * 2;      // 8 MB
  bf16_t* wqkvT  = (bf16_t*)ws; ws += (size_t)3 * EMB * EMB * 2;          // 6 MB
  bf16_t* wprojT = (bf16_t*)ws; ws += (size_t)EMB * EMB * 2;              // 2 MB
  bf16_t* qb     = (bf16_t*)ws; ws += (size_t)BATCH * NH * SEQ * HD * 2;  // 8 MB
  bf16_t* kb     = (bf16_t*)ws; ws += (size_t)BATCH * NH * SEQ * HD * 2;  // 8 MB
  bf16_t* vtb    = (bf16_t*)ws; ws += (size_t)BATCH * NH * HD * SEQ * 2;  // 8 MB
  bf16_t* attO   = (bf16_t*)ws; ws += (size_t)BATCH * SEQ * EMB * 2;      // 8 MB

  const int nx = BATCH * SEQ * EMB;
  cvt_f32_bf16<<<nx / 256, 256, 0, stream>>>(x, xb, nx);
  transpose_bf16<<<(3 * EMB * EMB) / 256, 256, 0, stream>>>(w_qkv, wqkvT, EMB, 3 * EMB);
  transpose_bf16<<<(EMB * EMB) / 256, 256, 0, stream>>>(w_proj, wprojT, EMB, EMB);

  qkv_gemm<<<dim3((BATCH * SEQ) / 128, (3 * EMB) / 128), 256, 0, stream>>>(xb, wqkvT, qb, kb, vtb);

  flash_attn<<<BATCH * NH * (SEQ / 64), 128, 0, stream>>>(qb, kb, vtb, attO);

  proj_gemm<<<dim3((BATCH * SEQ) / 128, EMB / 128), 256, 0, stream>>>(attO, wprojT, b_proj, out);
}
